// EuclidDistance_Assign_Module_21612275434145
// MI455X (gfx1250) — matrix operand order repaired and checked by reading, not compiled, not run on hardware
//
#include <hip/hip_runtime.h>
#include <hip/hip_bf16.h>

// ---------------------------------------------------------------------------
// Problem constants (from reference): B=8,D=16,H=32,W=32,C=512,K=256
// ---------------------------------------------------------------------------
#define M_TOT   131072        // B*D*H*W rows
#define CCH     512           // channels
#define KCL     256           // clusters
#define ALPHA_F 32.0f

// d_out flat offsets (floats), in reference return order
#define OFF_DIST   0LL
#define OFF_ASSIGN 33554432LL
#define OFF_CDIST  67108864LL
#define OFF_XREC   67174400LL
#define OFF_FEAT   134283264LL
#define OFF_LABEL  201392128LL

// d_ws byte offsets
#define WS_ROWSQ   0            // 131072 f32 = 524288 B
#define WS_CENSQ   524288       // 256 f32   = 1024 B
#define WS_CENB    525312       // 256x512 bf16 = 262144 B
#define WS_CENTB   787456       // 512x256 bf16 = 262144 B

typedef __attribute__((ext_vector_type(16))) __bf16       v16bf;
typedef __attribute__((ext_vector_type(8)))  float        v8f;
typedef __attribute__((ext_vector_type(4)))  float        f32x4;
typedef __attribute__((ext_vector_type(4)))  unsigned int u32x4;

union BfFrag { v16bf v; u32x4 u[2]; };

__device__ __forceinline__ unsigned short bf16bits(float f) {
  unsigned u = __builtin_bit_cast(unsigned, f);
  unsigned r = ((u >> 16) & 1u) + 0x7FFFu;   // round-to-nearest-even
  return (unsigned short)((u + r) >> 16);
}

// Single-instruction packed f32->bf16 convert (RNE), two values per VGPR.
__device__ __forceinline__ unsigned cvt_pk_bf16(float lo, float hi) {
  unsigned r;
  asm("v_cvt_pk_bf16_f32 %0, %1, %2" : "=v"(r) : "v"(lo), "v"(hi));
  return r;
}

// ---------------------------------------------------------------------------
// Kernel 1: LayerNorm. One wave per row; wave32 shuffle reductions.
// Writes feature (f32, d_out) and rowsq = sum(xn^2) (f32, ws).
// ---------------------------------------------------------------------------
__global__ __launch_bounds__(256) void ln_kernel(
    const float* __restrict__ x, const float* __restrict__ gamma,
    const float* __restrict__ beta, float* __restrict__ feature,
    float* __restrict__ rowsq)
{
  const int lane = threadIdx.x & 31;
  const int wave = threadIdx.x >> 5;
  const int row  = blockIdx.x * 8 + wave;
  const float* xr = x + (long long)row * CCH;

  f32x4 v[4];
  float s = 0.f, sq = 0.f;
  #pragma unroll
  for (int i = 0; i < 4; ++i) {
    v[i] = *(const f32x4*)(xr + i * 128 + lane * 4);
    s  += v[i].x + v[i].y + v[i].z + v[i].w;
    sq += v[i].x*v[i].x + v[i].y*v[i].y + v[i].z*v[i].z + v[i].w*v[i].w;
  }
  #pragma unroll
  for (int off = 16; off >= 1; off >>= 1) {
    s  += __shfl_xor(s,  off, 32);
    sq += __shfl_xor(sq, off, 32);
  }
  const float mu  = s * (1.0f / CCH);
  const float var = sq * (1.0f / CCH) - mu * mu;
  const float rs  = rsqrtf(var + 1e-5f);

  float nsq = 0.f;
  float* fr = feature + (long long)row * CCH;
  #pragma unroll
  for (int i = 0; i < 4; ++i) {
    const int c0 = i * 128 + lane * 4;
    f32x4 g = *(const f32x4*)(gamma + c0);
    f32x4 b = *(const f32x4*)(beta  + c0);
    f32x4 n;
    n.x = (v[i].x - mu) * rs * g.x + b.x;
    n.y = (v[i].y - mu) * rs * g.y + b.y;
    n.z = (v[i].z - mu) * rs * g.z + b.z;
    n.w = (v[i].w - mu) * rs * g.w + b.w;
    nsq += n.x*n.x + n.y*n.y + n.z*n.z + n.w*n.w;
    *(f32x4*)(fr + c0) = n;
  }
  #pragma unroll
  for (int off = 16; off >= 1; off >>= 1) nsq += __shfl_xor(nsq, off, 32);
  if (lane == 0) rowsq[row] = nsq;
}

// ---------------------------------------------------------------------------
// Kernel 2: center prep. One wave per center: bf16 copy, bf16 transpose, |c|^2
// ---------------------------------------------------------------------------
__global__ __launch_bounds__(32) void cen_prep(
    const float* __restrict__ cc, unsigned short* __restrict__ cenb,
    unsigned short* __restrict__ cenTb, float* __restrict__ censq)
{
  const int k = blockIdx.x, lane = threadIdx.x;
  const float* cr = cc + k * CCH;
  float sq = 0.f;
  #pragma unroll
  for (int j = 0; j < 4; ++j) {
    const int c0 = lane * 16 + j * 4;
    f32x4 v = *(const f32x4*)(cr + c0);
    sq += v.x*v.x + v.y*v.y + v.z*v.z + v.w*v.w;
    unsigned p0 = cvt_pk_bf16(v.x, v.y), p1 = cvt_pk_bf16(v.z, v.w);
    unsigned* dst = (unsigned*)(cenb + k * CCH + c0);
    dst[0] = p0; dst[1] = p1;
    cenTb[(c0 + 0) * KCL + k] = bf16bits(v.x);
    cenTb[(c0 + 1) * KCL + k] = bf16bits(v.y);
    cenTb[(c0 + 2) * KCL + k] = bf16bits(v.z);
    cenTb[(c0 + 3) * KCL + k] = bf16bits(v.w);
  }
  #pragma unroll
  for (int off = 16; off >= 1; off >>= 1) sq += __shfl_xor(sq, off, 32);
  if (lane == 0) censq[k] = sq;
}

// ---------------------------------------------------------------------------
// Kernel 3: cluster_dist [K,K] (tiny, f32, L2-resident)
// ---------------------------------------------------------------------------
__global__ __launch_bounds__(256) void cdist_kernel(
    const float* __restrict__ cc, const float* __restrict__ censq,
    float* __restrict__ out)
{
  const int i = blockIdx.x, j = threadIdx.x;
  const float* a = cc + i * CCH;
  const float* b = cc + j * CCH;
  float dot = 0.f;
  for (int c = 0; c < CCH; c += 4) {
    f32x4 va = *(const f32x4*)(a + c);
    f32x4 vb = *(const f32x4*)(b + c);
    dot += va.x*vb.x + va.y*vb.y + va.z*vb.z + va.w*vb.w;
  }
  float d2 = censq[i] + censq[j] - 2.0f * dot;
  out[i * KCL + j] = sqrtf(fmaxf(d2, 0.f));
}

// ---------------------------------------------------------------------------
// Kernel 4: fused dist + softmax/argmin + reconstruction GEMM.
// Block: 256 threads (8 waves), 64 rows. Wave w: row-tile rt=w&3, half h=w>>2.
// ---------------------------------------------------------------------------
__global__ __launch_bounds__(256) void fused_kernel(
    const float* __restrict__ feature, const float* __restrict__ rowsq,
    const unsigned short* __restrict__ cenb,
    const unsigned short* __restrict__ cenTb,
    const float* __restrict__ censq,
    float* __restrict__ xdist, float* __restrict__ xasg,
    float* __restrict__ xrec, float* __restrict__ label)
{
  extern __shared__ char smem[];
  float* s_dist = (float*)smem;                                  // 64*256 f32
  unsigned short* s_asg = (unsigned short*)(smem + 64 * 256 * 4); // 64*256 bf16

  const int tid  = threadIdx.x;
  const int lane = tid & 31;
  const int wave = tid >> 5;
  const int rt   = wave & 3;       // row tile 0..3
  const int h    = wave >> 2;      // column half 0..1
  const int m    = lane & 15;
  const int hi   = lane >> 4;      // lane-half select
  const long long blockRow = (long long)blockIdx.x * 64;

  const v8f vzero = {0.f,0.f,0.f,0.f,0.f,0.f,0.f,0.f};

  // ---------------- GEMM1: S[64,256] = Xn * C^T (bf16 WMMA) ----------------
  v8f acc[8];
  #pragma unroll
  for (int t = 0; t < 8; ++t) acc[t] = vzero;

  const float* aptr = feature + (blockRow + rt * 16 + m) * (long long)CCH;

  for (int ks = 0; ks < 16; ++ks) {
    const int kbase = ks * 32;
    __builtin_prefetch(cenb + (h * 128 + m) * CCH + kbase + 32, 0, 1);
    // A fragment: lanes 0-15 hold K {k..k+7, k+16..k+23}; lanes 16-31: +8
    const int aoff = kbase + hi * 8;
    f32x4 a0 = *(const f32x4*)(aptr + aoff);
    f32x4 a1 = *(const f32x4*)(aptr + aoff + 4);
    f32x4 a2 = *(const f32x4*)(aptr + aoff + 16);
    f32x4 a3 = *(const f32x4*)(aptr + aoff + 20);
    BfFrag af;
    af.u[0].x = cvt_pk_bf16(a0.x, a0.y); af.u[0].y = cvt_pk_bf16(a0.z, a0.w);
    af.u[0].z = cvt_pk_bf16(a1.x, a1.y); af.u[0].w = cvt_pk_bf16(a1.z, a1.w);
    af.u[1].x = cvt_pk_bf16(a2.x, a2.y); af.u[1].y = cvt_pk_bf16(a2.z, a2.w);
    af.u[1].z = cvt_pk_bf16(a3.x, a3.y); af.u[1].w = cvt_pk_bf16(a3.z, a3.w);
    #pragma unroll
    for (int ct = 0; ct < 8; ++ct) {
      const int n = h * 128 + ct * 16 + m;          // center index
      const unsigned short* bp = cenb + n * CCH + kbase + hi * 8;
      BfFrag bf;
      bf.u[0] = *(const u32x4*)(bp);
      bf.u[1] = *(const u32x4*)(bp + 16);
      acc[ct] = __builtin_amdgcn_wmma_f32_16x16x32_bf16(
          false, af.v, false, bf.v, (short)0, acc[ct], false, false);
    }
  }

  // dist = sqrt(max(|x|^2 + |c|^2 - 2S, 0)); write x_distance + LDS tile
  float rq[8];
  #pragma unroll
  for (int j = 0; j < 8; ++j)
    rq[j] = rowsq[blockRow + rt * 16 + hi * 8 + j];
  #pragma unroll
  for (int ct = 0; ct < 8; ++ct) {
    const int n = h * 128 + ct * 16 + m;
    const float cs = censq[n];
    #pragma unroll
    for (int j = 0; j < 8; ++j) {
      const int lr = rt * 16 + hi * 8 + j;
      float d2 = rq[j] + cs - 2.0f * acc[ct][j];
      float d  = sqrtf(fmaxf(d2, 0.f));
      xdist[(blockRow + lr) * 256 + n] = d;
      s_dist[lr * 256 + n] = d;
    }
  }
  __syncthreads();

  // ---------------- softmax(-alpha*d) + first-index argmin -----------------
  for (int q = 0; q < 8; ++q) {
    const int lr = wave * 8 + q;
    float v[8];
    #pragma unroll
    for (int i = 0; i < 8; ++i) v[i] = s_dist[lr * 256 + lane + 32 * i];
    float dmin = v[0]; int idx = lane;
    #pragma unroll
    for (int i = 1; i < 8; ++i) {
      const int c = lane + 32 * i;
      if (v[i] < dmin) { dmin = v[i]; idx = c; }
    }
    #pragma unroll
    for (int off = 16; off >= 1; off >>= 1) {
      float od = __shfl_xor(dmin, off, 32);
      int   oi = __shfl_xor(idx,  off, 32);
      if (od < dmin || (od == dmin && oi < idx)) { dmin = od; idx = oi; }
    }
    float e[8]; float ssum = 0.f;
    #pragma unroll
    for (int i = 0; i < 8; ++i) { e[i] = __expf(-ALPHA_F * (v[i] - dmin)); ssum += e[i]; }
    #pragma unroll
    for (int off = 16; off >= 1; off >>= 1) ssum += __shfl_xor(ssum, off, 32);
    const float inv = 1.0f / ssum;
    const long long grow = blockRow + lr;
    #pragma unroll
    for (int i = 0; i < 8; ++i) {
      const float a = e[i] * inv;
      xasg[grow * 256 + lane + 32 * i] = a;
      s_asg[lr * 256 + lane + 32 * i]  = bf16bits(a);
    }
    if (lane == 0) label[grow] = (float)idx;
  }
  __syncthreads();

  // -------------- GEMM2: x_rec[64,512] = Assign[64,256] * C ---------------
  for (int pass = 0; pass < 2; ++pass) {
    const int colbase = h * 256 + pass * 128;
    v8f racc[8];
    #pragma unroll
    for (int t = 0; t < 8; ++t) racc[t] = vzero;

    for (int ks = 0; ks < 8; ++ks) {
      const int kbase = ks * 32;
      const unsigned short* ap = s_asg + (rt * 16 + m) * 256 + kbase + hi * 8;
      BfFrag af;
      af.u[0] = *(const u32x4*)(ap);        // K {k..k+7}
      af.u[1] = *(const u32x4*)(ap + 16);   // K {k+16..k+23}
      #pragma unroll
      for (int ct = 0; ct < 8; ++ct) {
        const int n = colbase + ct * 16 + m;          // channel
        const unsigned short* bp = cenTb + n * KCL + kbase + hi * 8;
        BfFrag bf;
        bf.u[0] = *(const u32x4*)(bp);
        bf.u[1] = *(const u32x4*)(bp + 16);
        racc[ct] = __builtin_amdgcn_wmma_f32_16x16x32_bf16(
            false, af.v, false, bf.v, (short)0, racc[ct], false, false);
      }
    }
    #pragma unroll
    for (int ct = 0; ct < 8; ++ct) {
      const int col = colbase + ct * 16 + m;
      #pragma unroll
      for (int j = 0; j < 8; ++j) {
        const long long grow = blockRow + rt * 16 + hi * 8 + j;
        xrec[grow * CCH + col] = racc[ct][j];
      }
    }
  }
}

// ---------------------------------------------------------------------------
extern "C" void kernel_launch(void* const* d_in, const int* in_sizes, int n_in,
                              void* d_out, int out_size, void* d_ws, size_t ws_size,
                              hipStream_t stream) {
  const float* x     = (const float*)d_in[0];
  const float* cc    = (const float*)d_in[1];
  const float* gamma = (const float*)d_in[2];
  const float* beta  = (const float*)d_in[3];

  float* out = (float*)d_out;
  float* xdist = out + OFF_DIST;
  float* xasg  = out + OFF_ASSIGN;
  float* cdist = out + OFF_CDIST;
  float* xrec  = out + OFF_XREC;
  float* feat  = out + OFF_FEAT;
  float* label = out + OFF_LABEL;

  char* ws = (char*)d_ws;
  float*          rowsq = (float*)(ws + WS_ROWSQ);
  float*          censq = (float*)(ws + WS_CENSQ);
  unsigned short* cenb  = (unsigned short*)(ws + WS_CENB);
  unsigned short* cenTb = (unsigned short*)(ws + WS_CENTB);

  ln_kernel<<<M_TOT / 8, 256, 0, stream>>>(x, gamma, beta, feat, rowsq);
  cen_prep<<<KCL, 32, 0, stream>>>(cc, cenb, cenTb, censq);
  cdist_kernel<<<KCL, KCL, 0, stream>>>(cc, censq, cdist);
  fused_kernel<<<M_TOT / 64, 256, 96 * 1024, stream>>>(
      feat, rowsq, cenb, cenTb, censq, xdist, xasg, xrec, label);
}
